// Decoder_77326591197397
// MI455X (gfx1250) — compile-verified, matrix operand order checked by reading, not run on hardware
//
#include <hip/hip_runtime.h>
#include <math.h>

// ---------------- types ----------------
typedef __bf16 bf16;
typedef __bf16 bf16x8  __attribute__((ext_vector_type(8)));
typedef __bf16 bf16x16 __attribute__((ext_vector_type(16)));
typedef float  f32x8   __attribute__((ext_vector_type(8)));

#define DB 128     // batch
#define DS 1000    // encoder length
#define DT 256     // decoder steps
#define DV 30      // vocab
#define DE 256     // embed dim
#define DKV 256    // key/value dim
#define DH 512     // hidden

__device__ __forceinline__ float sigf(float x) { return 1.0f / (1.0f + __expf(-x)); }

__device__ __forceinline__ bf16x16 ldfrag(const bf16* p) {
  bf16x8 lo = *(const bf16x8*)(p);
  bf16x8 hi = *(const bf16x8*)(p + 16);
  return __builtin_shufflevector(lo, hi, 0,1,2,3,4,5,6,7,8,9,10,11,12,13,14,15);
}

// ---------------- setup kernels ----------------
__global__ void k_cvt(const float* __restrict__ in, bf16* __restrict__ out, long n) {
  long i = (long)blockIdx.x * blockDim.x + threadIdx.x;
  if (i < n) out[i] = (bf16)in[i];
}

// out[r, 0:ka] = Wa[r, :], out[r, ka:ka+kb] = Wb[r, :]  (bf16)
__global__ void k_cvt_cat(const float* __restrict__ Wa, const float* __restrict__ Wb,
                          bf16* __restrict__ out, int rows, int ka, int kb) {
  int kt = ka + kb;
  long i = (long)blockIdx.x * blockDim.x + threadIdx.x;
  if (i >= (long)rows * kt) return;
  int r = (int)(i / kt), k = (int)(i % kt);
  float v = (k < ka) ? Wa[(long)r * ka + k] : Wb[(long)r * kb + (k - ka)];
  out[i] = (bf16)v;
}

__global__ void k_bias_sum(const float* a1, const float* b1, float* o1,
                           const float* a2, const float* b2, float* o2, int n) {
  int i = blockIdx.x * blockDim.x + threadIdx.x;
  if (i >= n) return;
  o1[i] = a1[i] + b1[i];
  o2[i] = a2[i] + b2[i];
}

__global__ void k_zero(float* p, long n) {
  long i = (long)blockIdx.x * blockDim.x + threadIdx.x;
  if (i < n) p[i] = 0.0f;
}

// ctx0[b,k] = mean_s value[b,s,k]
__global__ void k_ctx0(const float* __restrict__ value, float* __restrict__ ctx) {
  int b = blockIdx.x, k = threadIdx.x;           // 128 blocks x 256 threads
  const float* vp = value + (size_t)b * DS * DKV + k;
  float s = 0.0f;
  for (int i = 0; i < DS; ++i) s += vp[(size_t)i * DKV];
  ctx[b * DKV + k] = s * (1.0f / DS);
}

// ---------------- per-step staging ----------------
// A1[b, :] = [ embed(y[t,b]) (256) | ctx (256) | h1 (512) ]  -> bf16, K=1024
__global__ void k_buildA1(const int* __restrict__ y, const float* __restrict__ emb,
                          const float* __restrict__ ctx, const float* __restrict__ h1,
                          bf16* __restrict__ A1, int t) {
  int idx = blockIdx.x * blockDim.x + threadIdx.x;   // 128*1024
  if (idx >= DB * 1024) return;
  int b = idx >> 10, col = idx & 1023;
  float v;
  if (col < DE) {
    int tok = y[t * DB + b];
    v = emb[tok * DE + col];
  } else if (col < DE + DKV) {
    v = ctx[b * DKV + (col - DE)];
  } else {
    v = h1[b * DH + (col - DE - DKV)];
  }
  A1[idx] = (bf16)v;
}

// ---------------- WMMA GEMM: C(128 x N) = A(128 x K) * Bw(N x K)^T + bias ----------
// Register-blocked: each wave computes a 32x64 output block (2 M-tiles x 4 N-tiles,
// 8 f32x8 accumulators). Per K-step: 6 fragment loads feed 8 v_wmma ops.
// Grid: waves = (M/32) * (N/64) = 4 * N/64; blocks = N/128 at 8 waves/block.
__global__ void k_wmma_gemm(const bf16* __restrict__ A, const bf16* __restrict__ Bw,
                            const float* __restrict__ bias, float* __restrict__ C,
                            int N, int K) {
  const int wave = threadIdx.x >> 5;
  const int lane = threadIdx.x & 31;
  int tile = blockIdx.x * (blockDim.x >> 5) + wave;
  int mIdx = tile & 3;            // 128/32 = 4 row blocks
  int nIdx = tile >> 2;
  if (nIdx >= (N >> 6)) return;

  const int r  = lane & 15;           // row within tile (A) / column (B)
  const int kh = (lane >> 4) << 3;    // 0 or 8: k-chunk base per lane half
  const int mBase = mIdx * 32;
  const int nBase = nIdx * 64;

  const bf16* pa0 = A  + (size_t)(mBase + r) * K + kh;
  const bf16* pa1 = A  + (size_t)(mBase + 16 + r) * K + kh;
  const bf16* pb0 = Bw + (size_t)(nBase + r) * K + kh;
  const bf16* pb1 = Bw + (size_t)(nBase + 16 + r) * K + kh;
  const bf16* pb2 = Bw + (size_t)(nBase + 32 + r) * K + kh;
  const bf16* pb3 = Bw + (size_t)(nBase + 48 + r) * K + kh;

  f32x8 acc00 = {}, acc01 = {}, acc02 = {}, acc03 = {};
  f32x8 acc10 = {}, acc11 = {}, acc12 = {}, acc13 = {};

  for (int kk = 0; kk < K; kk += 32) {
    bf16x16 a0 = ldfrag(pa0);
    bf16x16 a1 = ldfrag(pa1);
    bf16x16 b0 = ldfrag(pb0);
    bf16x16 b1 = ldfrag(pb1);
    bf16x16 b2 = ldfrag(pb2);
    bf16x16 b3 = ldfrag(pb3);
    __builtin_prefetch(pa0 + 128, 0, 1);   // global_prefetch_b8
    __builtin_prefetch(pb0 + 128, 0, 1);
    acc00 = __builtin_amdgcn_wmma_f32_16x16x32_bf16(false, a0, false, b0, (short)0, acc00, false, false);
    acc01 = __builtin_amdgcn_wmma_f32_16x16x32_bf16(false, a0, false, b1, (short)0, acc01, false, false);
    acc02 = __builtin_amdgcn_wmma_f32_16x16x32_bf16(false, a0, false, b2, (short)0, acc02, false, false);
    acc03 = __builtin_amdgcn_wmma_f32_16x16x32_bf16(false, a0, false, b3, (short)0, acc03, false, false);
    acc10 = __builtin_amdgcn_wmma_f32_16x16x32_bf16(false, a1, false, b0, (short)0, acc10, false, false);
    acc11 = __builtin_amdgcn_wmma_f32_16x16x32_bf16(false, a1, false, b1, (short)0, acc11, false, false);
    acc12 = __builtin_amdgcn_wmma_f32_16x16x32_bf16(false, a1, false, b2, (short)0, acc12, false, false);
    acc13 = __builtin_amdgcn_wmma_f32_16x16x32_bf16(false, a1, false, b3, (short)0, acc13, false, false);
    pa0 += 32; pa1 += 32;
    pb0 += 32; pb1 += 32; pb2 += 32; pb3 += 32;
  }

  // store: C layout per tile: VGPR i -> row = base + i (+8 for upper lane half), col = base + r
  const int rowoff = (lane >> 4) << 3;
  f32x8* accs[2][4] = {{&acc00, &acc01, &acc02, &acc03}, {&acc10, &acc11, &acc12, &acc13}};
#pragma unroll
  for (int ni = 0; ni < 4; ++ni) {
    int col = nBase + ni * 16 + r;
    float bb = bias ? bias[col] : 0.0f;
#pragma unroll
    for (int mi = 0; mi < 2; ++mi) {
      int rowbase = mBase + mi * 16 + rowoff;
      f32x8 v = *accs[mi][ni];
#pragma unroll
      for (int i = 0; i < 8; ++i)
        C[(size_t)(rowbase + i) * N + col] = v[i] + bb;
    }
  }
}

// ---------------- LSTM cells (elementwise) ----------------
// cell 1: also stages A2 = [c1_new | h2_prev] bf16
__global__ void k_lstm1(const float* __restrict__ gates, float* __restrict__ h1,
                        float* __restrict__ c1, const float* __restrict__ h2,
                        bf16* __restrict__ A2) {
  int idx = blockIdx.x * blockDim.x + threadIdx.x;   // 128*512
  if (idx >= DB * DH) return;
  int b = idx >> 9, j = idx & 511;
  const float* g = gates + (size_t)b * 4 * DH;
  float gi = g[j], gf = g[DH + j], gg = g[2 * DH + j], go = g[3 * DH + j];
  float c  = c1[idx];
  float cn = sigf(gf) * c + sigf(gi) * tanhf(gg);
  float hn = sigf(go) * tanhf(cn);
  c1[idx] = cn;
  h1[idx] = hn;
  A2[(size_t)b * 1024 + j]      = (bf16)cn;       // lstm2 input is c1_new
  A2[(size_t)b * 1024 + DH + j] = (bf16)h2[idx];  // previous h2
}

// cell 2: also stages c2 bf16 for the q GEMM
__global__ void k_lstm2(const float* __restrict__ gates, float* __restrict__ h2,
                        float* __restrict__ c2, bf16* __restrict__ c2b) {
  int idx = blockIdx.x * blockDim.x + threadIdx.x;
  if (idx >= DB * DH) return;
  int b = idx >> 9, j = idx & 511;
  const float* g = gates + (size_t)b * 4 * DH;
  float gi = g[j], gf = g[DH + j], gg = g[2 * DH + j], go = g[3 * DH + j];
  float c  = c2[idx];
  float cn = sigf(gf) * c + sigf(gi) * tanhf(gg);
  float hn = sigf(go) * tanhf(cn);
  c2[idx]  = cn;
  h2[idx]  = hn;
  c2b[idx] = (bf16)cn;
}

// ---------------- attention: one workgroup (256 thr) per batch row ----------------
template <int BF16KV>
__global__ void k_attn(const void* __restrict__ encp, const void* __restrict__ valp,
                       const float* __restrict__ q, float* __restrict__ ctx,
                       float* __restrict__ attn0, int t) {
  __shared__ float sq[DKV];
  __shared__ float se[DS];
  __shared__ float sr[256];
  int b = blockIdx.x, tid = threadIdx.x;
  sq[tid] = q[b * DKV + tid];
  __syncthreads();

  const float rs = 0.031622776601683794f;   // 1/sqrt(1000)
  float lmax = -3.4e38f;
  for (int s = tid; s < DS; s += 256) {
    float acc = 0.0f;
    if (BF16KV) {
      const bf16* e = (const bf16*)encp + ((size_t)b * DS + s) * DKV;
#pragma unroll 4
      for (int k = 0; k < DKV; k += 8) {
        bf16x8 v = *(const bf16x8*)(e + k);
#pragma unroll
        for (int j = 0; j < 8; ++j) acc += (float)v[j] * sq[k + j];
      }
    } else {
      const float* e = (const float*)encp + ((size_t)b * DS + s) * DKV;
#pragma unroll 8
      for (int k = 0; k < DKV; ++k) acc += e[k] * sq[k];
    }
    float en = (acc + 1e-9f) * rs;
    se[s] = en;
    lmax = fmaxf(lmax, en);
  }
  sr[tid] = lmax; __syncthreads();
  for (int o = 128; o > 0; o >>= 1) {
    if (tid < o) sr[tid] = fmaxf(sr[tid], sr[tid + o]);
    __syncthreads();
  }
  float m = sr[0];
  __syncthreads();

  float lsum = 0.0f;
  for (int s = tid; s < DS; s += 256) {
    float e = __expf(se[s] - m);
    se[s] = e;
    lsum += e;
  }
  sr[tid] = lsum; __syncthreads();
  for (int o = 128; o > 0; o >>= 1) {
    if (tid < o) sr[tid] += sr[tid + o];
    __syncthreads();
  }
  float inv = 1.0f / sr[0];
  __syncthreads();
  for (int s = tid; s < DS; s += 256) se[s] *= inv;
  __syncthreads();

  // ctx[b,k] = sum_s attn[s] * value[b,s,k]; 256 threads cover all k (coalesced rows)
  float acc = 0.0f;
  if (BF16KV) {
    const bf16* v = (const bf16*)valp + (size_t)b * DS * DKV + tid;
    for (int s = 0; s < DS; ++s) acc += se[s] * (float)v[(size_t)s * DKV];
  } else {
    const float* v = (const float*)valp + (size_t)b * DS * DKV + tid;
    for (int s = 0; s < DS; ++s) acc += se[s] * v[(size_t)s * DKV];
  }
  ctx[b * DKV + tid] = acc;

  if (b == 0)
    for (int s = tid; s < DS; s += 256) attn0[(size_t)t * DS + s] = se[s];
}

// ---------------- output projection: pred = [ctx|q] @ Wc^T + bc ----------------
__global__ void k_pred(const float* __restrict__ ctx, const float* __restrict__ q,
                       const float* __restrict__ Wc, const float* __restrict__ bc,
                       float* __restrict__ preds, int t) {
  int gid = blockIdx.x * blockDim.x + threadIdx.x;   // 128*30
  if (gid >= DB * DV) return;
  int b = gid / DV, v = gid % DV;
  const float* w  = Wc + (size_t)v * (2 * DKV);
  const float* cp = ctx + b * DKV;
  const float* qp = q + b * DKV;
  float acc = bc[v];
#pragma unroll 8
  for (int k = 0; k < DKV; ++k) acc += cp[k] * w[k];
#pragma unroll 8
  for (int k = 0; k < DKV; ++k) acc += qp[k] * w[DKV + k];
  preds[(size_t)b * DT * DV + t * DV + v] = acc;
}

// ---------------- host ----------------
static inline int gb(long n, int bs) { return (int)((n + bs - 1) / bs); }

extern "C" void kernel_launch(void* const* d_in, const int* in_sizes, int n_in,
                              void* d_out, int out_size, void* d_ws, size_t ws_size,
                              hipStream_t stream) {
  const float* enc_key = (const float*)d_in[0];
  const float* value   = (const float*)d_in[1];
  const int*   y       = (const int*)d_in[2];
  /* d_in[3] encoder_len is unused (reference discards the mask) */
  const float* emb     = (const float*)d_in[4];
  const float* W_ih1   = (const float*)d_in[5];
  const float* b_ih1   = (const float*)d_in[6];
  const float* W_hh1   = (const float*)d_in[7];
  const float* b_hh1   = (const float*)d_in[8];
  const float* W_ih2   = (const float*)d_in[9];
  const float* b_ih2   = (const float*)d_in[10];
  const float* W_hh2   = (const float*)d_in[11];
  const float* b_hh2   = (const float*)d_in[12];
  const float* Wq      = (const float*)d_in[13];
  const float* bq      = (const float*)d_in[14];
  const float* Wc      = (const float*)d_in[15];
  const float* bc      = (const float*)d_in[16];

  float* preds = (float*)d_out;                       // [B, T*V]
  float* attn0 = preds + (size_t)DB * DT * DV;        // [T, S, 1]

  // ---- workspace carve-out ----
  char* ws = (char*)d_ws;
  size_t off = 0;
  auto alloc = [&](size_t bytes) -> char* {
    char* p = ws + off;
    off = (off + bytes + 255) & ~(size_t)255;
    return p;
  };
  bf16*  W1b   = (bf16*)alloc((size_t)4 * DH * 1024 * 2);   // [2048,1024]
  bf16*  W2b   = (bf16*)alloc((size_t)4 * DH * 1024 * 2);
  bf16*  Wqb   = (bf16*)alloc((size_t)DKV * DH * 2);
  float* b1s   = (float*)alloc((size_t)4 * DH * 4);
  float* b2s   = (float*)alloc((size_t)4 * DH * 4);
  float* state = (float*)alloc((size_t)4 * DB * DH * 4);    // h1,c1,h2,c2
  float* h1 = state, *c1 = state + DB * DH, *h2 = state + 2 * DB * DH, *c2 = state + 3 * DB * DH;
  float* ctx    = (float*)alloc((size_t)DB * DKV * 4);
  bf16*  A1     = (bf16*)alloc((size_t)DB * 1024 * 2);
  bf16*  A2     = (bf16*)alloc((size_t)DB * 1024 * 2);
  bf16*  c2b    = (bf16*)alloc((size_t)DB * DH * 2);
  float* gates1 = (float*)alloc((size_t)DB * 4 * DH * 4);
  float* gates2 = (float*)alloc((size_t)DB * 4 * DH * 4);
  float* qbuf   = (float*)alloc((size_t)DB * DKV * 4);

  // enc/value bf16 staging (131 MB) only if workspace allows — fits in 192 MB L2
  const size_t kvElems = (size_t)DB * DS * DKV;
  bool useBf16 = (ws_size >= off + 2 * kvElems * 2 + 512);
  bf16* encb = nullptr; bf16* valb = nullptr;
  if (useBf16) {
    encb = (bf16*)alloc(kvElems * 2);
    valb = (bf16*)alloc(kvElems * 2);
  }

  // ---- one-time setup (runs every call; deterministic) ----
  k_cvt_cat<<<gb((long)4 * DH * 1024, 256), 256, 0, stream>>>(W_ih1, W_hh1, W1b, 4 * DH, DH, DH);
  k_cvt_cat<<<gb((long)4 * DH * 1024, 256), 256, 0, stream>>>(W_ih2, W_hh2, W2b, 4 * DH, DH, DH);
  k_cvt<<<gb((long)DKV * DH, 256), 256, 0, stream>>>(Wq, Wqb, (long)DKV * DH);
  k_bias_sum<<<gb(4 * DH, 256), 256, 0, stream>>>(b_ih1, b_hh1, b1s, b_ih2, b_hh2, b2s, 4 * DH);
  k_zero<<<gb((long)4 * DB * DH, 256), 256, 0, stream>>>(state, (long)4 * DB * DH);
  k_ctx0<<<DB, DKV, 0, stream>>>(value, ctx);
  if (useBf16) {
    k_cvt<<<gb((long)kvElems, 256), 256, 0, stream>>>(enc_key, encb, (long)kvElems);
    k_cvt<<<gb((long)kvElems, 256), 256, 0, stream>>>(value, valb, (long)kvElems);
  }

  // ---- recurrent loop (graph-captured) ----
  for (int t = 0; t < DT; ++t) {
    k_buildA1<<<gb((long)DB * 1024, 256), 256, 0, stream>>>(y, emb, ctx, h1, A1, t);
    // gates1 = [x|h1] @ [W_ih1|W_hh1]^T + (b_ih1+b_hh1): N=2048, K=1024 -> 16 blocks
    k_wmma_gemm<<<(4 * DH) / 128, 256, 0, stream>>>(A1, W1b, b1s, gates1, 4 * DH, 1024);
    k_lstm1<<<gb((long)DB * DH, 256), 256, 0, stream>>>(gates1, h1, c1, h2, A2);
    // gates2 = [c1|h2] @ [W_ih2|W_hh2]^T + (b_ih2+b_hh2)
    k_wmma_gemm<<<(4 * DH) / 128, 256, 0, stream>>>(A2, W2b, b2s, gates2, 4 * DH, 1024);
    k_lstm2<<<gb((long)DB * DH, 256), 256, 0, stream>>>(gates2, h2, c2, c2b);
    // q = c2 @ Wq^T + bq: N=256, K=512 -> 2 blocks
    k_wmma_gemm<<<DKV / 128, 256, 0, stream>>>(c2b, Wqb, bq, qbuf, DKV, DH);
    if (useBf16)
      k_attn<1><<<DB, 256, 0, stream>>>(encb, valb, qbuf, ctx, attn0, t);
    else
      k_attn<0><<<DB, 256, 0, stream>>>(enc_key, value, qbuf, ctx, attn0, t);
    k_pred<<<gb((long)DB * DV, 256), 256, 0, stream>>>(ctx, qbuf, Wc, bc, preds, t);
  }
}
